// SelfAttention_15204184228345
// MI455X (gfx1250) — compile-verified, matrix operand order checked by reading, not run on hardware
//
#include <hip/hip_runtime.h>
#include <stdint.h>

// ---------------------------------------------------------------------------
// MI455X (gfx1250) fused self-attention forward, bf16 WMMA + f32 accumulate.
// B=2 S=2048 E=1024 H=16 HD=64 ROT=64 (full-dim RoPE). Compute-bound
// (~69 GFLOP vs ~150MB HBM): V_WMMA_F32_16X16X32_BF16 everywhere,
// flash-attention streaming so S x S scores never touch HBM.
// Tile staging uses GLOBAL_LOAD_ASYNC_TO_LDS_B128 (ASYNCcnt-tracked direct
// global->LDS path, CDNA5 ISA) instead of the
// global_load -> VGPR -> ds_store round-trip.
// ---------------------------------------------------------------------------

typedef __attribute__((ext_vector_type(16))) __bf16 v16bf;
typedef __attribute__((ext_vector_type(8)))  float  v8f;

#define B_    2
#define S_    2048
#define E_    1024
#define H_    16
#define HD_   64
#define M_TOT (B_ * S_)    // 4096
#define N_QKV (3 * E_)     // 3072

// Exact parameter types per hipcc diagnostic: pointer to 16B int vector,
// global (addrspace 1) source, LDS (addrspace 3) destination.
typedef int v4i __attribute__((vector_size(4 * sizeof(int))));
typedef __attribute__((address_space(1))) v4i* gptr_v4i;
typedef __attribute__((address_space(3))) v4i* lptr_v4i;

// Direct global->LDS async copy, 16 bytes per lane (ASYNCcnt tracked).
__device__ __forceinline__ void async_cp16(void* lds, const void* g) {
  __builtin_amdgcn_global_load_async_to_lds_b128((gptr_v4i)(void*)g,
                                                 (lptr_v4i)lds, 0, 0);
}
// Copy one 32B slot (16 bf16) as two b128 async transfers.
__device__ __forceinline__ void async_cp32(void* lds, const void* g) {
  async_cp16(lds, g);
  async_cp16((char*)lds + 16, (const char*)g + 16);
}

__device__ __forceinline__ __bf16 f2bf(float f) {
  union { float f; uint32_t u; } c; c.f = f;
  uint32_t r = (c.u + 0x7FFFu + ((c.u >> 16) & 1u)) >> 16;  // RNE truncate
  union { uint16_t s; __bf16 h; } o; o.s = (uint16_t)r;
  return o.h;
}

// 16-bit A-fragment layout (ISA 7.12.2): lane m = lane%16 holds row m; VGPR v
// holds K pair kk = 2v + (v>=4 ? 8:0) + 8*(lane/16). B-fragments are the
// mirror (lane = column N), so a K-major row read serves both A and W^T-as-B.
__device__ __forceinline__ v16bf frag_row(const __bf16* base, int stride,
                                          int m, int half) {
  v16bf a;
  const __bf16* p = base + m * stride + 8 * half;
#pragma unroll
  for (int v = 0; v < 8; ++v) {
    const int kk = 2 * v + ((v >= 4) ? 8 : 0);
    union { uint32_t u; __bf16 h[2]; } cv;
    cv.u = *(const uint32_t*)(p + kk);        // 4B-aligned pair load
    a[2 * v] = cv.h[0];
    a[2 * v + 1] = cv.h[1];
  }
  return a;
}

// B-fragment when the K dimension is the *row* of the staged tile (V tile):
// lane n reads column n at rows kk.
__device__ __forceinline__ v16bf frag_col(const __bf16* base, int stride,
                                          int n, int half) {
  v16bf b;
#pragma unroll
  for (int v = 0; v < 8; ++v) {
    const int kk = 2 * v + ((v >= 4) ? 8 : 0) + 8 * half;
    b[2 * v]     = base[kk * stride + n];
    b[2 * v + 1] = base[(kk + 1) * stride + n];
  }
  return b;
}

__device__ __forceinline__ v8f wmma_bf16(v16bf a, v16bf b, v8f c) {
  // D = A(16x32) * B(32x16) + C, f32 accumulate.
  return __builtin_amdgcn_wmma_f32_16x16x32_bf16(false, a, false, b,
                                                 (short)0, c, false, false);
}

// ---------------------------------------------------------------------------
__global__ void cvt_f32_bf16(const float* __restrict__ src,
                             __bf16* __restrict__ dst, int n) {
  int i = blockIdx.x * blockDim.x + threadIdx.x;
  const int st = gridDim.x * blockDim.x;
  for (; i < n; i += st) dst[i] = f2bf(src[i]);
}

// ---------------------------------------------------------------------------
// QKV GEMM (M=4096, N=3072, K=1024) + bias + RoPE, scatter to (B,H,S,HD).
#define TM  128
#define TN  128
#define TK  32
#define LDT 34   // 17-dword row stride: conflict-free 16-row parallel reads

__global__ __launch_bounds__(256)
void qkv_rope_kernel(const __bf16* __restrict__ xb,
                     const __bf16* __restrict__ wq,
                     const float* __restrict__ bias,
                     __bf16* __restrict__ qh, __bf16* __restrict__ kh,
                     __bf16* __restrict__ vh) {
  __shared__ __bf16 Asm[TM * LDT];
  __shared__ __bf16 Bsm[TN * LDT];
  const int tid = threadIdx.x;
  const int lane = tid & 31, wid = tid >> 5;
  const int lane16 = lane & 15, half = (lane >> 4) & 1;
  const int wm = wid & 3, wn = wid >> 2;   // 4x2 waves, wave tile 32(M)x64(N)
  const int m0 = blockIdx.x * TM;
  const int n0 = blockIdx.y * TN;
  const int sr = tid >> 1;                 // staging row 0..127
  const int sc = (tid & 1) * 16;           // staging col chunk

  v8f acc[2][4] = {};

  for (int k0 = 0; k0 < E_; k0 += TK) {
    __syncthreads();   // previous tiles fully consumed
    async_cp32(&Asm[sr * LDT + sc], xb + (size_t)(m0 + sr) * E_ + k0 + sc);
    async_cp32(&Bsm[sr * LDT + sc], wq + (size_t)(n0 + sr) * E_ + k0 + sc);
    __builtin_amdgcn_s_wait_asynccnt(0);
    __syncthreads();
    if (k0 + TK < E_) {
      __builtin_prefetch(xb + (size_t)(m0 + sr) * E_ + k0 + TK + sc, 0, 1);
      __builtin_prefetch(wq + (size_t)(n0 + sr) * E_ + k0 + TK + sc, 0, 1);
    }
    v16bf af[2], bfr[4];
#pragma unroll
    for (int mt = 0; mt < 2; ++mt)
      af[mt] = frag_row(&Asm[(wm * 32 + mt * 16) * LDT], LDT, lane16, half);
#pragma unroll
    for (int nt = 0; nt < 4; ++nt)
      bfr[nt] = frag_row(&Bsm[(wn * 64 + nt * 16) * LDT], LDT, lane16, half);
#pragma unroll
    for (int mt = 0; mt < 2; ++mt)
#pragma unroll
      for (int nt = 0; nt < 4; ++nt)
        acc[mt][nt] = wmma_bf16(af[mt], bfr[nt], acc[mt][nt]);
  }

  // Epilogue: bias + RoPE (pair partner = adjacent fragment lane) + scatter.
#pragma unroll
  for (int mt = 0; mt < 2; ++mt) {
#pragma unroll
    for (int nt = 0; nt < 4; ++nt) {
      const int n = n0 + wn * 64 + nt * 16 + lane16;
      const int sec = n >> 10;                // 0=q 1=k 2=v (uniform per frag)
      const int hn = n & 1023, h = hn >> 6, d = hn & 63;
      const float bv = bias[n];
      __bf16* dst = (sec == 0) ? qh : (sec == 1) ? kh : vh;
#pragma unroll
      for (int i = 0; i < 8; ++i) {
        const int r = m0 + wm * 32 + mt * 16 + i + 8 * half;
        const int b = r >> 11, s = r & (S_ - 1);
        float val = acc[mt][nt][i] + bv;
        if (sec < 2) {
          const float partner = __shfl_xor(val, 1, 32);
          // inv_freq = BASE^(-(d&~1)/64); ln(1e4)/64 = 0.14391156831212787
          const float inv = __expf(-(float)(d & ~1) * 0.14391156831212787f);
          float sn, cs;
          __sincosf((float)s * inv, &sn, &cs);
          val = (d & 1) ? (val * cs + partner * sn)
                        : (val * cs - partner * sn);
        }
        dst[((size_t)((b * H_ + h) * S_ + s)) * HD_ + d] = f2bf(val);
      }
    }
  }
}

// ---------------------------------------------------------------------------
// Flash attention: block = 4 waves, 64 q rows per block, one (b,h) per
// blockIdx.y. K/V streamed through LDS in 64-key tiles; online softmax.
#define LKV 66   // 33-dword padded row stride

__global__ __launch_bounds__(128)
void attn_kernel(const __bf16* __restrict__ qh, const __bf16* __restrict__ kh,
                 const __bf16* __restrict__ vh, const int* __restrict__ mask,
                 __bf16* __restrict__ ctx) {
  __shared__ __bf16 Kt[64 * LKV];
  __shared__ __bf16 Vt[64 * LKV];
  __shared__ __bf16 Pt[4 * 16 * LKV];
  const int tid = threadIdx.x;
  const int lane = tid & 31, w = tid >> 5;
  const int lane16 = lane & 15, half = (lane >> 4) & 1;
  const int bh = blockIdx.y;
  const int b = bh >> 4, h = bh & 15;
  const int q0 = blockIdx.x * 64 + w * 16;
  const __bf16* qbase = qh + ((size_t)bh * S_ + q0) * HD_;
  const __bf16* kbase = kh + (size_t)bh * S_ * HD_;
  const __bf16* vbase = vh + (size_t)bh * S_ * HD_;
  __bf16* Pw = &Pt[w * 16 * LKV];

  v16bf qa[2];
#pragma unroll
  for (int ks = 0; ks < 2; ++ks)
    qa[ks] = frag_row(qbase + 32 * ks, HD_, lane16, half);

  v8f o[4] = {};
  float m[8], l[8];
#pragma unroll
  for (int i = 0; i < 8; ++i) { m[i] = -3.0e38f; l[i] = 0.f; }

  for (int j0 = 0; j0 < S_; j0 += 64) {
#pragma unroll
    for (int it = 0; it < 2; ++it) {          // stage 64x64 K and V tiles
      const int idx = tid + it * 128;
      const int r = idx >> 2, c = (idx & 3) * 16;
      async_cp32(&Kt[r * LKV + c], kbase + (size_t)(j0 + r) * HD_ + c);
      async_cp32(&Vt[r * LKV + c], vbase + (size_t)(j0 + r) * HD_ + c);
    }
    __builtin_amdgcn_s_wait_asynccnt(0);
    __syncthreads();
    if (j0 + 64 < S_) {
      __builtin_prefetch(kbase + (size_t)(j0 + 64) * HD_ + tid * 32, 0, 1);
      __builtin_prefetch(vbase + (size_t)(j0 + 64) * HD_ + tid * 32, 0, 1);
    }

    // scores: q(16x64) . K^T(64x64) -> 4 tiles, 2 k-steps each
    v8f sc[4] = {};
#pragma unroll
    for (int nt = 0; nt < 4; ++nt)
#pragma unroll
      for (int ks = 0; ks < 2; ++ks) {
        const v16bf kb = frag_row(&Kt[nt * 16 * LKV + 32 * ks], LKV, lane16, half);
        sc[nt] = wmma_bf16(qa[ks], kb, sc[nt]);
      }

    // scale 1/sqrt(64) + key mask
#pragma unroll
    for (int nt = 0; nt < 4; ++nt) {
      const int mv = mask[b * S_ + j0 + nt * 16 + lane16];
      const float mb = mv ? 0.f : -1.0e30f;
#pragma unroll
      for (int i = 0; i < 8; ++i) sc[nt][i] = sc[nt][i] * 0.125f + mb;
    }

    // online softmax; rows live at (VGPR i, half) -> 16-wide lane reductions
#pragma unroll
    for (int i = 0; i < 8; ++i) {
      float v = fmaxf(fmaxf(sc[0][i], sc[1][i]), fmaxf(sc[2][i], sc[3][i]));
      v = fmaxf(v, __shfl_xor(v, 1, 16));
      v = fmaxf(v, __shfl_xor(v, 2, 16));
      v = fmaxf(v, __shfl_xor(v, 4, 16));
      v = fmaxf(v, __shfl_xor(v, 8, 16));
      const float mn = fmaxf(m[i], v);
      const float corr = __expf(m[i] - mn);
      m[i] = mn;
      l[i] *= corr;
#pragma unroll
      for (int od = 0; od < 4; ++od) o[od][i] *= corr;
      float rs = 0.f;
#pragma unroll
      for (int nt = 0; nt < 4; ++nt) {
        const float p = __expf(sc[nt][i] - mn);
        rs += p;
        Pw[(i + 8 * half) * LKV + nt * 16 + lane16] = f2bf(p);  // C->A reshape
      }
      rs += __shfl_xor(rs, 1, 16);
      rs += __shfl_xor(rs, 2, 16);
      rs += __shfl_xor(rs, 4, 16);
      rs += __shfl_xor(rs, 8, 16);
      l[i] += rs;
    }
    asm volatile("" ::: "memory");  // P is wave-private; DS is in-order/wave

    // O += P(16x64) . V(64x64)
#pragma unroll
    for (int ks = 0; ks < 2; ++ks) {
      const v16bf pa = frag_row(Pw + 32 * ks, LKV, lane16, half);
#pragma unroll
      for (int od = 0; od < 4; ++od) {
        const v16bf vb = frag_col(&Vt[32 * ks * LKV], LKV, od * 16 + lane16, half);
        o[od] = wmma_bf16(pa, vb, o[od]);
      }
    }
    __syncthreads();
  }

  // finalize: ctx row-major (B*S, E), col = h*64 + d
#pragma unroll
  for (int i = 0; i < 8; ++i) {
    const float inv = 1.f / fmaxf(l[i], 1e-37f);
    const int s = q0 + i + 8 * half;
    const size_t rowoff = ((size_t)(b * S_ + s)) * E_ + h * HD_;
#pragma unroll
    for (int od = 0; od < 4; ++od)
      ctx[rowoff + od * 16 + lane16] = f2bf(o[od][i] * inv);
  }
}

// ---------------------------------------------------------------------------
// Output GEMM (M=4096, N=1024, K=1024) + bias, fp32 out.
__global__ __launch_bounds__(256)
void out_gemm_kernel(const __bf16* __restrict__ ctx,
                     const __bf16* __restrict__ wo,
                     const float* __restrict__ bias,
                     float* __restrict__ out) {
  __shared__ __bf16 Asm[TM * LDT];
  __shared__ __bf16 Bsm[TN * LDT];
  const int tid = threadIdx.x;
  const int lane = tid & 31, wid = tid >> 5;
  const int lane16 = lane & 15, half = (lane >> 4) & 1;
  const int wm = wid & 3, wn = wid >> 2;
  const int m0 = blockIdx.x * TM;
  const int n0 = blockIdx.y * TN;
  const int sr = tid >> 1;
  const int sc = (tid & 1) * 16;

  v8f acc[2][4] = {};

  for (int k0 = 0; k0 < E_; k0 += TK) {
    __syncthreads();
    async_cp32(&Asm[sr * LDT + sc], ctx + (size_t)(m0 + sr) * E_ + k0 + sc);
    async_cp32(&Bsm[sr * LDT + sc], wo + (size_t)(n0 + sr) * E_ + k0 + sc);
    __builtin_amdgcn_s_wait_asynccnt(0);
    __syncthreads();
    if (k0 + TK < E_) {
      __builtin_prefetch(ctx + (size_t)(m0 + sr) * E_ + k0 + TK + sc, 0, 1);
      __builtin_prefetch(wo + (size_t)(n0 + sr) * E_ + k0 + TK + sc, 0, 1);
    }
    v16bf af[2], bfr[4];
#pragma unroll
    for (int mt = 0; mt < 2; ++mt)
      af[mt] = frag_row(&Asm[(wm * 32 + mt * 16) * LDT], LDT, lane16, half);
#pragma unroll
    for (int nt = 0; nt < 4; ++nt)
      bfr[nt] = frag_row(&Bsm[(wn * 64 + nt * 16) * LDT], LDT, lane16, half);
#pragma unroll
    for (int mt = 0; mt < 2; ++mt)
#pragma unroll
      for (int nt = 0; nt < 4; ++nt)
        acc[mt][nt] = wmma_bf16(af[mt], bfr[nt], acc[mt][nt]);
  }

#pragma unroll
  for (int mt = 0; mt < 2; ++mt)
#pragma unroll
    for (int nt = 0; nt < 4; ++nt) {
      const int n = n0 + wn * 64 + nt * 16 + lane16;
      const float bv = bias[n];
#pragma unroll
      for (int i = 0; i < 8; ++i) {
        const int r = m0 + wm * 32 + mt * 16 + i + 8 * half;
        out[(size_t)r * E_ + n] = acc[mt][nt][i] + bv;
      }
    }
}

// ---------------------------------------------------------------------------
extern "C" void kernel_launch(void* const* d_in, const int* in_sizes, int n_in,
                              void* d_out, int out_size, void* d_ws,
                              size_t ws_size, hipStream_t stream) {
  const float* x     = (const float*)d_in[0];
  const int*   mask  = (const int*)d_in[1];
  const float* qkv_w = (const float*)d_in[2];
  const float* qkv_b = (const float*)d_in[3];
  const float* out_w = (const float*)d_in[4];
  const float* out_b = (const float*)d_in[5];
  float* out = (float*)d_out;

  char* ws = (char*)d_ws;
  const size_t MB = 1024 * 1024;
  __bf16* xb    = (__bf16*)(ws + 0 * MB);    // 4096x1024   (8 MB)
  __bf16* wqkvb = (__bf16*)(ws + 8 * MB);    // 3072x1024   (6 MB)
  __bf16* woutb = (__bf16*)(ws + 14 * MB);   // 1024x1024   (2 MB)
  __bf16* qh    = (__bf16*)(ws + 16 * MB);   // (B,H,S,HD)  (8 MB)
  __bf16* kh    = (__bf16*)(ws + 24 * MB);   // (B,H,S,HD)  (8 MB)
  __bf16* vh    = (__bf16*)(ws + 32 * MB);   // (B,H,S,HD)  (8 MB)
  __bf16* ctx   = (__bf16*)(ws + 40 * MB);   // 4096x1024   (8 MB)

  cvt_f32_bf16<<<1024, 256, 0, stream>>>(x, xb, M_TOT * E_);
  cvt_f32_bf16<<<1024, 256, 0, stream>>>(qkv_w, wqkvb, N_QKV * E_);
  cvt_f32_bf16<<<512, 256, 0, stream>>>(out_w, woutb, E_ * E_);

  dim3 g1(M_TOT / TM, N_QKV / TN);   // 32 x 24
  qkv_rope_kernel<<<g1, 256, 0, stream>>>(xb, wqkvb, qkv_b, qh, kh, vh);

  dim3 g2(S_ / 64, B_ * H_);         // 32 x 32
  attn_kernel<<<g2, 128, 0, stream>>>(qh, kh, vh, mask, ctx);

  dim3 g3(M_TOT / TM, E_ / TN);      // 32 x 8
  out_gemm_kernel<<<g3, 256, 0, stream>>>(ctx, woutb, out_b, out);
}
